// LatentODE_33809982554318
// MI455X (gfx1250) — matrix operand order repaired and checked by reading, not compiled, not run on hardware
//
#include <hip/hip_runtime.h>
#include <hip/hip_bf16.h>
#include <math.h>

#define B_  8192
#define T_  100
#define D_  44
#define NH_ 64
#define L_  8

typedef __attribute__((ext_vector_type(16))) _Float16 v16h;
typedef __attribute__((ext_vector_type(8)))  _Float16 v8h;
typedef __attribute__((ext_vector_type(8)))  float    v8f;

// ---------------------------------------------------------------------------
// CDNA5 WMMA helpers (wave32, V_WMMA_F32_16X16X32_F16).
// Fragment layouts per CDNA5 ISA §7.12.2.
// ---------------------------------------------------------------------------

__device__ __forceinline__ v8f wmma_f16(v16h a, v16h b, v8f c) {
  // 8 args: (neg_a, A, neg_b, B, c_mod, C, reuse_a, reuse_b)
  return __builtin_amdgcn_wmma_f32_16x16x32_f16(false, a, false, b, (short)0, c,
                                                false, false);
}

// A-matrix 16x32 f16 fragment from a row-major [16 x ldk] f16 LDS buffer.
// lanes 0-15: row m=lane, K={0..7} in halves 0..7, K={16..23} in halves 8..15
// lanes 16-31: row m=lane-16, K={8..15} and K={24..31}.
// Offsets chosen so both 8-half groups are 16B-aligned ds_load_b128.
__device__ __forceinline__ v16h ldsA(const _Float16* act, int ldk, int kBase,
                                     int lane) {
  int m  = lane & 15;
  int hi = (lane >> 4) << 3;              // 0 or 8
  const _Float16* p = act + m * ldk + kBase + hi;
  v8h lo = *(const v8h*)(p);
  v8h hh = *(const v8h*)(p + 16);
  v16h a;
#pragma unroll
  for (int i = 0; i < 8; ++i) { a[i] = lo[i]; a[i + 8] = hh[i]; }
  return a;
}

// B-matrix 32x16 f16 fragment from row-major [K x ldn] f16 LDS weights.
// halves 0..7 hold rows k0..k0+7, halves 8..15 rows k0+16..k0+23, col n = nBase + (lane&15),
// k0 = kBase (lanes 0-15) or kBase+8 (lanes 16-31).
__device__ __forceinline__ v16h ldsB(const _Float16* W, int ldn, int kBase,
                                     int nBase, int lane) {
  int n  = nBase + (lane & 15);
  int k0 = kBase + ((lane >> 4) << 3);
  v16h b;
#pragma unroll
  for (int v = 0; v < 8; ++v) {
    b[v]     = W[(k0 + v) * ldn + n];
    b[v + 8] = W[(k0 + 16 + v) * ldn + n];
  }
  return b;
}

// Branch-free fast tanh on the native v_exp_f32 / v_rcp_f32 path:
// tanh(x) = sign(x) * (1 - e)/(1 + e),  e = exp(-2|x|)
__device__ __forceinline__ float fast_tanh(float x) {
  float e = __expf(-2.0f * fabsf(x));
  float r = (1.0f - e) * __builtin_amdgcn_rcpf(1.0f + e);
  return copysignf(r, x);
}

__device__ __forceinline__ float elu(float v) {
  return v > 0.f ? v : __expf(v) - 1.f;
}

// wave32 sum reduction + one atomic per wave
__device__ __forceinline__ void wave_atomic_add(float* dst, float v, int lane) {
#pragma unroll
  for (int off = 16; off > 0; off >>= 1) v += __shfl_down(v, off, 32);
  if (lane == 0) atomicAdd(dst, v);
}

// ---------------------------------------------------------------------------
// Kernel 0: init loss with the constant Gaussian term:
//   mean_b( sum_{t,d} 0.5*(log2pi + noise_logvar) ) = T*D*0.5*(log2pi+nlv)
// ---------------------------------------------------------------------------
__global__ void k_init(float* loss) {
  if (threadIdx.x == 0 && blockIdx.x == 0) {
    const float log2pi = 1.837877066409345f;
    const float nlv    = 2.0f * logf(0.3f);
    loss[0] = (float)(T_ * D_) * 0.5f * (log2pi + nlv);
  }
}

// ---------------------------------------------------------------------------
// Kernel 1: reverse RNN encoder + VAE head + KL.
// Block = 128 threads (4 waves), 16 batch rows.
// A buffer layout (K padded to 128): cols [0,44)=masked x_t, [44,64)=0 pad,
// [64,128)=h. Wi2h staged to match (rows 44..63 zero), so every ldsA is
// 16B aligned. Wave w owns output N-tile [16w,16w+16): 4 WMMAs per step.
// ---------------------------------------------------------------------------
__global__ __launch_bounds__(128) void k_rnn(
    const float* __restrict__ x, const float* __restrict__ mask,
    const float* __restrict__ eps, const float* __restrict__ Wi2h,
    const float* __restrict__ bi2h, const float* __restrict__ Wh2o,
    const float* __restrict__ bh2o, float* __restrict__ predz,
    float* __restrict__ loss) {
  __shared__ __align__(16) _Float16 sW[128 * 64];   // padded Wi2h
  __shared__ __align__(16) _Float16 sW2[64 * 16];   // Wh2o
  __shared__ __align__(16) _Float16 sA[16 * 128];   // [x | pad | h]
  __shared__ float sB1[64];
  __shared__ float sB2[16];
  __shared__ float sOut[16 * 16];

  const int tid = threadIdx.x;
  const int lane = tid & 31, w = tid >> 5;
  const long b0 = (long)blockIdx.x * 16;

  for (int e = tid; e < 128 * 64; e += 128) {
    int k = e >> 6, n = e & 63;
    float v = 0.f;
    if (k < 44)       v = Wi2h[k * 64 + n];           // x rows
    else if (k >= 64) v = Wi2h[(k - 20) * 64 + n];    // h rows (44..107)
    sW[e] = (_Float16)v;
  }
  for (int e = tid; e < 64 * 16; e += 128) sW2[e] = (_Float16)Wh2o[e];
  for (int e = tid; e < 16 * 128; e += 128) sA[e] = (_Float16)0.f;  // h0=0
  if (tid < 64) sB1[tid] = bi2h[tid];
  if (tid < 16) sB2[tid] = bh2o[tid];
  __syncthreads();

  // Preload this wave's 4 K-chunk weight fragments (stay in VGPRs).
  v16h bw[4];
#pragma unroll
  for (int kc = 0; kc < 4; ++kc) bw[kc] = ldsB(sW, 64, kc * 32, w * 16, lane);

  const int nCol = w * 16 + (lane & 15);
  const int mOff = (lane >> 4) * 8;
  const float bn = sB1[nCol];

  for (int t = 0; t < T_; ++t) {
    const int ti = T_ - 1 - t;  // reverse time
    for (int e = tid; e < 16 * D_; e += 128) {
      int r = e / D_, c2 = e % D_;
      long g = ((b0 + r) * T_ + ti) * D_ + c2;
      sA[r * 128 + c2] = (_Float16)(x[g] * mask[g]);
    }
    __syncthreads();
    v8f c = {};
#pragma unroll
    for (int kc = 0; kc < 4; ++kc) {
      v16h a = ldsA(sA, 128, kc * 32, lane);
      c = wmma_f16(a, bw[kc], c);
    }
    __syncthreads();  // all reads of old h done before overwrite
#pragma unroll
    for (int r = 0; r < 8; ++r)
      sA[(mOff + r) * 128 + 64 + nCol] = (_Float16)fast_tanh(c[r] + bn);
  }
  __syncthreads();

  // Head: out(16x16) = h(16x64) @ Wh2o(64x16) + b, wave 0 only.
  if (w == 0) {
    v8f c = {};
#pragma unroll
    for (int kc = 0; kc < 2; ++kc) {
      v16h a  = ldsA(sA, 128, 64 + kc * 32, lane);
      v16h bb = ldsB(sW2, 16, kc * 32, 0, lane);
      c = wmma_f16(a, bb, c);
    }
    int n = lane & 15;
#pragma unroll
    for (int r = 0; r < 8; ++r) sOut[(mOff + r) * 16 + n] = c[r] + sB2[n];
  }
  __syncthreads();

  // z0 = eps*exp(0.5*logvar)+mean ; KL accumulation. Thread = (row, j).
  const int row = tid >> 3, j = tid & 7;
  float mean = sOut[row * 16 + j];
  float lv   = sOut[row * 16 + 8 + j];
  long  gb   = b0 + row;
  float ep   = eps[gb * 8 + j];
  float z0   = ep * __expf(0.5f * lv) + mean;
  predz[(gb * T_ + 0) * 8 + j] = z0;
  float kl = -0.5f * lv + 0.5f * (__expf(lv) + mean * mean) - 0.5f;
  wave_atomic_add(loss, kl * (1.0f / (float)B_), lane);
}

// ---------------------------------------------------------------------------
// Kernel 2: RK4 ODE integration.  Block = 128 threads, 16 batch rows.
// f(z): z(16x8, K-pad 32) @ Wf1 -> elu -> @ Wf2(64x64) -> elu -> @ Wf3
// (N-pad 8->16). 14 WMMAs per f-eval per block, 56 per RK4 step, 99 steps.
// Each thread owns one (row, j) element of z.
// ---------------------------------------------------------------------------
__global__ __launch_bounds__(128) void k_ode(
    const float* __restrict__ Wf1, const float* __restrict__ bf1,
    const float* __restrict__ Wf2, const float* __restrict__ bf2,
    const float* __restrict__ Wf3, const float* __restrict__ bf3,
    float* __restrict__ predz) {
  __shared__ __align__(16) _Float16 sW1[32 * 64];   // K-padded Wf1
  __shared__ __align__(16) _Float16 sW2[64 * 64];
  __shared__ __align__(16) _Float16 sW3[64 * 16];   // N-padded Wf3
  __shared__ __align__(16) _Float16 sZ[16 * 32];
  __shared__ __align__(16) _Float16 sH1[16 * 64];
  __shared__ __align__(16) _Float16 sH2[16 * 64];
  __shared__ float sF[16 * 8];
  __shared__ float sb1[64], sb2[64], sb3[8];

  const int tid = threadIdx.x, lane = tid & 31, w = tid >> 5;
  const long b0 = (long)blockIdx.x * 16;

  for (int e = tid; e < 32 * 64; e += 128) {
    int k = e >> 6, n = e & 63;
    sW1[e] = (_Float16)(k < 8 ? Wf1[k * 64 + n] : 0.f);
  }
  for (int e = tid; e < 64 * 64; e += 128) sW2[e] = (_Float16)Wf2[e];
  for (int e = tid; e < 64 * 16; e += 128) {
    int k = e >> 4, n = e & 15;
    sW3[e] = (_Float16)(n < 8 ? Wf3[k * 8 + n] : 0.f);
  }
  for (int e = tid; e < 16 * 32; e += 128) sZ[e] = (_Float16)0.f;  // K pad
  if (tid < 64) { sb1[tid] = bf1[tid]; sb2[tid] = bf2[tid]; }
  if (tid < 8) sb3[tid] = bf3[tid];
  __syncthreads();

  v16h b1f = ldsB(sW1, 64, 0, w * 16, lane);
  v16h b2f[2] = { ldsB(sW2, 64, 0, w * 16, lane),
                  ldsB(sW2, 64, 32, w * 16, lane) };
  v16h b3f[2] = { ldsB(sW3, 16, 0, 0, lane), ldsB(sW3, 16, 32, 0, lane) };

  const int nCol = w * 16 + (lane & 15);
  const int mOff = (lane >> 4) * 8;
  const float b1n = sb1[nCol], b2n = sb2[nCol];

  const int row = tid >> 3, j = tid & 7;
  float z = predz[(b0 + row) * T_ * 8 + j];  // z0 from encoder
  const float dt = 1.0f / (float)(T_ - 1);

  auto feval = [&](float zin) -> float {
    sZ[row * 32 + j] = (_Float16)zin;
    __syncthreads();
    {  // layer 1: 8(->32) -> 64, elu
      v16h a = ldsA(sZ, 32, 0, lane);
      v8f c = {};
      c = wmma_f16(a, b1f, c);
#pragma unroll
      for (int r = 0; r < 8; ++r)
        sH1[(mOff + r) * 64 + nCol] = (_Float16)elu(c[r] + b1n);
    }
    __syncthreads();
    {  // layer 2: 64 -> 64, elu
      v8f c = {};
#pragma unroll
      for (int kc = 0; kc < 2; ++kc) {
        v16h a = ldsA(sH1, 64, kc * 32, lane);
        c = wmma_f16(a, b2f[kc], c);
      }
#pragma unroll
      for (int r = 0; r < 8; ++r)
        sH2[(mOff + r) * 64 + nCol] = (_Float16)elu(c[r] + b2n);
    }
    __syncthreads();
    if (w == 0) {  // layer 3: 64 -> 8 (N-padded 16)
      v8f c = {};
#pragma unroll
      for (int kc = 0; kc < 2; ++kc) {
        v16h a = ldsA(sH2, 64, kc * 32, lane);
        c = wmma_f16(a, b3f[kc], c);
      }
      int n = lane & 15;
      if (n < 8) {
#pragma unroll
        for (int r = 0; r < 8; ++r) sF[(mOff + r) * 8 + n] = c[r] + sb3[n];
      }
    }
    __syncthreads();
    return sF[row * 8 + j];
  };

  for (int s = 0; s < T_ - 1; ++s) {
    float k1 = feval(z);
    float k2 = feval(z + 0.5f * dt * k1);
    float k3 = feval(z + 0.5f * dt * k2);
    float k4 = feval(z + dt * k3);
    z = z + dt * (1.0f / 6.0f) * (k1 + 2.f * k2 + 2.f * k3 + k4);
    predz[((b0 + row) * T_ + s + 1) * 8 + j] = z;
  }
}

// ---------------------------------------------------------------------------
// Kernel 3: decoder + masked Gaussian NLL.  Block = 16 (b,t) rows.
// relu(z @ Wd1 + b) @ Wd2 (N-pad 44->48) + b, then residual reduction.
// ---------------------------------------------------------------------------
__global__ __launch_bounds__(128) void k_dec(
    const float* __restrict__ x, const float* __restrict__ mask,
    const float* __restrict__ Wd1, const float* __restrict__ bd1,
    const float* __restrict__ Wd2, const float* __restrict__ bd2,
    const float* __restrict__ predz, float* __restrict__ loss) {
  __shared__ __align__(16) _Float16 sW1[32 * 64];  // K-padded Wd1
  __shared__ __align__(16) _Float16 sW2[64 * 48];  // N-padded Wd2
  __shared__ __align__(16) _Float16 sZ[16 * 32];
  __shared__ __align__(16) _Float16 sH[16 * 64];
  __shared__ float sP[16 * 48];
  __shared__ float sb1[64], sb2[48];

  const int tid = threadIdx.x, lane = tid & 31, w = tid >> 5;
  const long r0 = (long)blockIdx.x * 16;  // flattened (b*T + t) row base

  for (int e = tid; e < 32 * 64; e += 128) {
    int k = e >> 6, n = e & 63;
    sW1[e] = (_Float16)(k < 8 ? Wd1[k * 64 + n] : 0.f);
  }
  for (int e = tid; e < 64 * 48; e += 128) {
    int k = e / 48, n = e % 48;
    sW2[e] = (_Float16)(n < 44 ? Wd2[k * 44 + n] : 0.f);
  }
  for (int e = tid; e < 16 * 32; e += 128) {
    int r = e >> 5, c = e & 31;
    sZ[e] = (_Float16)(c < 8 ? predz[(r0 + r) * 8 + c] : 0.f);
  }
  if (tid < 64) sb1[tid] = bd1[tid];
  if (tid < 48) sb2[tid] = (tid < 44 ? bd2[tid] : 0.f);
  __syncthreads();

  const int nCol = w * 16 + (lane & 15);
  const int mOff = (lane >> 4) * 8;
  {  // layer 1: 8(->32) -> 64, relu
    v16h a  = ldsA(sZ, 32, 0, lane);
    v16h bb = ldsB(sW1, 64, 0, w * 16, lane);
    v8f c = {};
    c = wmma_f16(a, bb, c);
    float b1n = sb1[nCol];
#pragma unroll
    for (int r = 0; r < 8; ++r) {
      float v = c[r] + b1n;
      sH[(mOff + r) * 64 + nCol] = (_Float16)(v > 0.f ? v : 0.f);
    }
  }
  __syncthreads();
  if (w < 3) {  // layer 2: 64 -> 44 (padded 48), 3 N-tiles x 2 K-chunks
    v8f c = {};
#pragma unroll
    for (int kc = 0; kc < 2; ++kc) {
      v16h a  = ldsA(sH, 64, kc * 32, lane);
      v16h bb = ldsB(sW2, 48, kc * 32, w * 16, lane);
      c = wmma_f16(a, bb, c);
    }
    int n = w * 16 + (lane & 15);
#pragma unroll
    for (int r = 0; r < 8; ++r) sP[(mOff + r) * 48 + n] = c[r] + sb2[n];
  }
  __syncthreads();

  const float inv_var = 1.0f / 0.09f;  // exp(-noise_logvar), sigma=0.3
  float local = 0.f;
  for (int e = tid; e < 16 * D_; e += 128) {
    int r = e / D_, c = e % D_;
    long g = (r0 + r) * D_ + c;
    float xv = x[g], mv = mask[g];
    float mx  = xv * mv;
    float upd = (mx != 0.f) ? mx : sP[r * 48 + c];
    float d   = xv - upd;
    local += 0.5f * d * d * inv_var;
  }
  wave_atomic_add(loss, local * (1.0f / (float)B_), lane);
}

// ---------------------------------------------------------------------------
extern "C" void kernel_launch(void* const* d_in, const int* in_sizes, int n_in,
                              void* d_out, int out_size, void* d_ws,
                              size_t ws_size, hipStream_t stream) {
  const float* x    = (const float*)d_in[0];
  const float* mask = (const float*)d_in[1];
  const float* eps  = (const float*)d_in[2];
  const float* Wi2h = (const float*)d_in[3];
  const float* bi2h = (const float*)d_in[4];
  const float* Wh2o = (const float*)d_in[5];
  const float* bh2o = (const float*)d_in[6];
  const float* Wf1  = (const float*)d_in[7];
  const float* bf1  = (const float*)d_in[8];
  const float* Wf2  = (const float*)d_in[9];
  const float* bf2  = (const float*)d_in[10];
  const float* Wf3  = (const float*)d_in[11];
  const float* bf3  = (const float*)d_in[12];
  const float* Wd1  = (const float*)d_in[13];
  const float* bd1  = (const float*)d_in[14];
  const float* Wd2  = (const float*)d_in[15];
  const float* bd2  = (const float*)d_in[16];
  float* out   = (float*)d_out;
  float* predz = (float*)d_ws;  // B*T*L floats = 26.2 MB scratch

  k_init<<<1, 32, 0, stream>>>(out);
  k_rnn<<<B_ / 16, 128, 0, stream>>>(x, mask, eps, Wi2h, bi2h, Wh2o, bh2o,
                                     predz, out);
  k_ode<<<B_ / 16, 128, 0, stream>>>(Wf1, bf1, Wf2, bf2, Wf3, bf3, predz);
  k_dec<<<(B_ * T_) / 16, 128, 0, stream>>>(x, mask, Wd1, bd1, Wd2, bd2, predz,
                                            out);
}
